// CausalSelfAttention_20899310862440
// MI455X (gfx1250) — matrix operand order repaired and checked by reading, not compiled, not run on hardware
//
#include <hip/hip_runtime.h>
#include <hip/hip_bf16.h>

// CausalSelfAttention forward for MI455X (gfx1250), bf16 WMMA pipeline.
// B=2, T=2048, C=1024, NH=16, HD=64.

typedef __bf16 bf16_t;
typedef __attribute__((ext_vector_type(16))) __bf16 v16bf;
typedef __attribute__((ext_vector_type(8)))  __bf16 v8bf;
typedef __attribute__((ext_vector_type(4)))  __bf16 v4bf;
typedef __attribute__((ext_vector_type(8)))  float  v8f;
typedef __attribute__((ext_vector_type(4)))  int    v4i;

// Address-space-qualified pointer types for the async-LDS builtin.
typedef v4i __attribute__((address_space(1)))* as1_v4i_p;
typedef v4i __attribute__((address_space(3)))* as3_v4i_p;

#define B_DIM  2
#define T_DIM  2048
#define C_DIM  1024
#define NH_DIM 16
#define HD_DIM 64

#if __has_builtin(__builtin_amdgcn_global_load_async_to_lds_b128)
#define USE_ASYNC_LDS 1
static __device__ __forceinline__ void wait_asynccnt0() {
#if __has_builtin(__builtin_amdgcn_s_wait_asynccnt)
  __builtin_amdgcn_s_wait_asynccnt(0);
#else
  asm volatile("s_wait_asynccnt 0x0" ::: "memory");
#endif
}
static __device__ __forceinline__ void async_copy_b128(const bf16_t* g,
                                                       bf16_t* l) {
  __builtin_amdgcn_global_load_async_to_lds_b128((as1_v4i_p)g, (as3_v4i_p)l, 0, 0);
}
#endif

static __device__ __forceinline__ bf16_t to_bf16(float f) {
  unsigned u = __builtin_bit_cast(unsigned, f);
  u += 0x7FFFu + ((u >> 16) & 1u);               // round-to-nearest-even
  unsigned short h = (unsigned short)(u >> 16);
  return __builtin_bit_cast(bf16_t, h);
}

// ---------------------------------------------------------------- cast f32->bf16
__global__ void cast4_f32_bf16(const float* __restrict__ in,
                               bf16_t* __restrict__ out, int n4) {
  int i = blockIdx.x * blockDim.x + threadIdx.x;
  if (i < n4) {
    float4 f = ((const float4*)in)[i];
    v4bf o;
    o[0] = to_bf16(f.x); o[1] = to_bf16(f.y);
    o[2] = to_bf16(f.z); o[3] = to_bf16(f.w);
    ((v4bf*)out)[i] = o;
  }
}

// ------------------------------------------------- transpose V to [B,NH,HD,T]
__global__ void vtrans(const bf16_t* __restrict__ qkv, bf16_t* __restrict__ vt) {
  int t   = blockIdx.x * blockDim.x + threadIdx.x;   // 0..T-1
  int bhd = blockIdx.y;                              // b*NH*HD + h*HD + d
  int d   = bhd % HD_DIM;
  int bh  = bhd / HD_DIM;
  int h   = bh % NH_DIM;
  int b   = bh / NH_DIM;
  vt[(size_t)bhd * T_DIM + t] =
      qkv[((size_t)(b * T_DIM + t)) * (3 * C_DIM) + 2 * C_DIM + h * HD_DIM + d];
}

// ---------------------------------------------------------------- WMMA GEMM
// C[M,N] = A[M,K] * B[K,N], row-major, bf16 in, f32 accumulate.
// Block tile 128x128, 8 waves of 32x64, K-step 32. A staged async to LDS.
template <bool OUT_BF16>
__global__ __launch_bounds__(256) void gemm_bf16_wmma(
    const bf16_t* __restrict__ A, const bf16_t* __restrict__ Bm,
    void* __restrict__ Cout, int M, int N, int K) {
  __shared__ bf16_t As[128 * 32];   // [m][k]
  __shared__ bf16_t Bs[128 * 32];   // transposed: [n][k]

  const int tid      = threadIdx.x;
  const int wid      = tid >> 5;
  const int lane     = tid & 31;
  const int laneHalf = lane >> 4;   // 0 or 1
  const int lane16   = lane & 15;
  const int m0       = blockIdx.y * 128;
  const int n0       = blockIdx.x * 128;
  const int waveM    = (wid & 3) * 32;
  const int waveN    = (wid >> 2) * 64;

  v8f acc[2][4] = {};

  for (int kk = 0; kk < K; kk += 32) {
    // ---- stage A tile 128x32 (two 16B chunks per thread)
    {
      int c = tid;
      int r = c >> 2, cc = (c & 3) * 8;
#ifdef USE_ASYNC_LDS
      async_copy_b128(&A[(size_t)(m0 + r) * K + kk + cc], &As[r * 32 + cc]);
      c = tid + 256; r = c >> 2; cc = (c & 3) * 8;
      async_copy_b128(&A[(size_t)(m0 + r) * K + kk + cc], &As[r * 32 + cc]);
#else
      *(uint4*)&As[r * 32 + cc] = *(const uint4*)&A[(size_t)(m0 + r) * K + kk + cc];
      c = tid + 256; r = c >> 2; cc = (c & 3) * 8;
      *(uint4*)&As[r * 32 + cc] = *(const uint4*)&A[(size_t)(m0 + r) * K + kk + cc];
#endif
    }
    // ---- stage B tile 32x128, transposed into Bs[n][k] (16 elems/thread)
    {
      int k  = tid >> 3;
      int nc = (tid & 7) * 16;
      v8bf b0 = *(const v8bf*)&Bm[(size_t)(kk + k) * N + n0 + nc];
      v8bf b1 = *(const v8bf*)&Bm[(size_t)(kk + k) * N + n0 + nc + 8];
#pragma unroll
      for (int j = 0; j < 8; ++j) {
        Bs[(nc + j) * 32 + k]     = b0[j];
        Bs[(nc + 8 + j) * 32 + k] = b1[j];
      }
    }
#ifdef USE_ASYNC_LDS
    wait_asynccnt0();
#endif
    __syncthreads();

    // ---- fragments
    v16bf a[2], b[4];
#pragma unroll
    for (int i = 0; i < 2; ++i) {
      const bf16_t* ap = &As[(waveM + i * 16 + lane16) * 32 + laneHalf * 16];
      v8bf lo = *(const v8bf*)ap;
      v8bf hi = *(const v8bf*)(ap + 8);
#pragma unroll
      for (int e = 0; e < 8; ++e) { a[i][e] = lo[e]; a[i][8 + e] = hi[e]; }
    }
#pragma unroll
    for (int j = 0; j < 4; ++j)
      b[j] = *(const v16bf*)&Bs[(waveN + j * 16 + lane16) * 32 + laneHalf * 16];

#pragma unroll
    for (int i = 0; i < 2; ++i)
#pragma unroll
      for (int j = 0; j < 4; ++j)
        acc[i][j] = __builtin_amdgcn_wmma_f32_16x16x32_bf16(
            false, a[i], false, b[j], (short)0, acc[i][j], false, false);
    __syncthreads();
  }

  // ---- store
#pragma unroll
  for (int i = 0; i < 2; ++i) {
#pragma unroll
    for (int j = 0; j < 4; ++j) {
#pragma unroll
      for (int v = 0; v < 8; ++v) {
        int row = m0 + waveM + i * 16 + laneHalf * 8 + v;
        int col = n0 + waveN + j * 16 + lane16;
        if (OUT_BF16)
          ((bf16_t*)Cout)[(size_t)row * N + col] = to_bf16(acc[i][j][v]);
        else
          ((float*)Cout)[(size_t)row * N + col] = acc[i][j][v];
      }
    }
  }
}

// ------------------------------------------------------------ flash attention
// One wave per 16-query tile of one (b,h). Online softmax (base-2), 32-key blocks.
__global__ __launch_bounds__(256) void flash_attn(
    const bf16_t* __restrict__ qkv, const bf16_t* __restrict__ vt,
    bf16_t* __restrict__ yout) {
  __shared__ bf16_t pLds[8][16 * 32];   // per-wave P transpose staging

  const float SCALE_LOG2E = 0.125f * 1.4426950408889634f;  // HD^-0.5 * log2(e)
  const int tid      = threadIdx.x;
  const int wid      = tid >> 5;
  const int lane     = tid & 31;
  const int laneHalf = lane >> 4;
  const int lane16   = lane & 15;

  const int qbPerHead = T_DIM / 128;                 // 16
  const int head   = blockIdx.x / qbPerHead;
  const int qBlock = blockIdx.x % qbPerHead;
  const int h = head % NH_DIM;
  const int b = head / NH_DIM;
  const int qTile = qBlock * 8 + wid;
  const int q0 = qTile * 16;

  const size_t rs = 3 * C_DIM;                       // qkv row stride
  const bf16_t* qbase = qkv + (size_t)b * T_DIM * rs + (size_t)h * HD_DIM;
  const bf16_t* kbase = qbase + C_DIM;
  const bf16_t* vtb   = vt + ((size_t)(b * NH_DIM + h) * HD_DIM) * T_DIM;

  // ---- load Q fragments (16 x 64 = two K-chunks of 32)
  v16bf aQ[2];
#pragma unroll
  for (int c = 0; c < 2; ++c) {
    const bf16_t* qp = qbase + (size_t)(q0 + lane16) * rs + c * 32 + laneHalf * 16;
    v8bf lo = *(const v8bf*)qp;
    v8bf hi = *(const v8bf*)(qp + 8);
#pragma unroll
    for (int e = 0; e < 8; ++e) { aQ[c][e] = lo[e]; aQ[c][8 + e] = hi[e]; }
  }

  v8f   yacc[4] = {};
  float mrow[8], lrow[8];
#pragma unroll
  for (int v = 0; v < 8; ++v) { mrow[v] = -3.0e38f; lrow[v] = 0.0f; }

  const int nkb = (q0 + 16 + 31) / 32;               // causal key-block count
  for (int kb = 0; kb < nkb; ++kb) {
    const int j0 = kb * 32;

    // ---- S = Q @ K^T for 32 keys (two 16-col tiles)
    v8f sacc[2];
#pragma unroll
    for (int nt = 0; nt < 2; ++nt) {
      const bf16_t* kp = kbase + (size_t)(j0 + nt * 16 + lane16) * rs + laneHalf * 16;
      v16bf bk0 = *(const v16bf*)kp;
      v16bf bk1 = *(const v16bf*)(kp + 32);
      v8f z = {};
      z = __builtin_amdgcn_wmma_f32_16x16x32_bf16(false, aQ[0], false, bk0,
                                                  (short)0, z, false, false);
      sacc[nt] = __builtin_amdgcn_wmma_f32_16x16x32_bf16(
          false, aQ[1], false, bk1, (short)0, z, false, false);
    }

    // ---- online softmax in base 2 (rows live across 16 consecutive lanes)
#pragma unroll
    for (int v = 0; v < 8; ++v) {
      const int row = q0 + laneHalf * 8 + v;
      float s0 = sacc[0][v] * SCALE_LOG2E;
      float s1 = sacc[1][v] * SCALE_LOG2E;
      if (j0 + lane16      > row) s0 = -3.0e38f;     // causal mask
      if (j0 + 16 + lane16 > row) s1 = -3.0e38f;
      float rm = fmaxf(s0, s1);
      rm = fmaxf(rm, __shfl_xor(rm, 1, 32));
      rm = fmaxf(rm, __shfl_xor(rm, 2, 32));
      rm = fmaxf(rm, __shfl_xor(rm, 4, 32));
      rm = fmaxf(rm, __shfl_xor(rm, 8, 32));
      float mnew  = fmaxf(mrow[v], rm);
      float alpha = exp2f(mrow[v] - mnew);
      mrow[v] = mnew;
      float p0 = exp2f(s0 - mnew);
      float p1 = exp2f(s1 - mnew);
      float ps = p0 + p1;
      ps += __shfl_xor(ps, 1, 32);
      ps += __shfl_xor(ps, 2, 32);
      ps += __shfl_xor(ps, 4, 32);
      ps += __shfl_xor(ps, 8, 32);
      lrow[v] = lrow[v] * alpha + ps;
#pragma unroll
      for (int ct = 0; ct < 4; ++ct) yacc[ct][v] *= alpha;
      pLds[wid][(laneHalf * 8 + v) * 32 + lane16]      = to_bf16(p0);
      pLds[wid][(laneHalf * 8 + v) * 32 + 16 + lane16] = to_bf16(p1);
    }

    // ---- P (C-layout) -> A-fragment via per-wave LDS transpose
    v16bf aP;
    {
      const bf16_t* pp = &pLds[wid][lane16 * 32 + laneHalf * 16];
      v8bf lo = *(const v8bf*)pp;
      v8bf hi = *(const v8bf*)(pp + 8);
#pragma unroll
      for (int e = 0; e < 8; ++e) { aP[e] = lo[e]; aP[8 + e] = hi[e]; }
    }

    // ---- Y += P @ V  (V pre-transposed: contiguous along keys)
#pragma unroll
    for (int ct = 0; ct < 4; ++ct) {
      v16bf bv = *(const v16bf*)&vtb[(size_t)(ct * 16 + lane16) * T_DIM +
                                     j0 + laneHalf * 16];
      yacc[ct] = __builtin_amdgcn_wmma_f32_16x16x32_bf16(
          false, aP, false, bv, (short)0, yacc[ct], false, false);
    }
  }

  // ---- normalize and store as bf16 into [B,T,C]
#pragma unroll
  for (int v = 0; v < 8; ++v) {
    const float inv = 1.0f / lrow[v];
    const int row = q0 + laneHalf * 8 + v;
#pragma unroll
    for (int ct = 0; ct < 4; ++ct) {
      int col = h * HD_DIM + ct * 16 + lane16;
      yout[((size_t)b * T_DIM + row) * C_DIM + col] = to_bf16(yacc[ct][v] * inv);
    }
  }
}

// ---------------------------------------------------------------- launcher
extern "C" void kernel_launch(void* const* d_in, const int* in_sizes, int n_in,
                              void* d_out, int out_size, void* d_ws, size_t ws_size,
                              hipStream_t stream) {
  (void)in_sizes; (void)n_in; (void)out_size; (void)ws_size;
  const float* x     = (const float*)d_in[0];   // [B,T,C]
  const float* wqkv  = (const float*)d_in[1];   // [C,3C]
  const float* wproj = (const float*)d_in[2];   // [C,C]
  float* out = (float*)d_out;                   // [B,T,C]

  const size_t nX    = (size_t)B_DIM * T_DIM * C_DIM;       // 4 Mi
  const size_t nWqkv = (size_t)C_DIM * 3 * C_DIM;           // 3 Mi
  const size_t nWprj = (size_t)C_DIM * C_DIM;               // 1 Mi
  const size_t nQkv  = (size_t)B_DIM * T_DIM * 3 * C_DIM;   // 12 Mi

  bf16_t* xb     = (bf16_t*)d_ws;          // ws layout (bf16): 56 MB total
  bf16_t* wqkvb  = xb + nX;
  bf16_t* wprojb = wqkvb + nWqkv;
  bf16_t* qkvb   = wprojb + nWprj;
  bf16_t* vtb    = qkvb + nQkv;            // [B,NH,HD,T]
  bf16_t* yb     = vtb + nX;               // attention output [B,T,C]

  // 1) casts to bf16
  cast4_f32_bf16<<<dim3((unsigned)(nX / 4 / 256)), 256, 0, stream>>>(x, xb, (int)(nX / 4));
  cast4_f32_bf16<<<dim3((unsigned)(nWqkv / 4 / 256)), 256, 0, stream>>>(wqkv, wqkvb, (int)(nWqkv / 4));
  cast4_f32_bf16<<<dim3((unsigned)(nWprj / 4 / 256)), 256, 0, stream>>>(wproj, wprojb, (int)(nWprj / 4));

  // 2) QKV projection: [4096,1024] @ [1024,3072] -> bf16 qkv
  gemm_bf16_wmma<true><<<dim3(3 * C_DIM / 128, B_DIM * T_DIM / 128), 256, 0, stream>>>(
      xb, wqkvb, qkvb, B_DIM * T_DIM, 3 * C_DIM, C_DIM);

  // 3) transpose V to [B,NH,HD,T]
  vtrans<<<dim3(T_DIM / 256, B_DIM * NH_DIM * HD_DIM), 256, 0, stream>>>(qkvb, vtb);

  // 4) causal flash attention -> bf16 y [B,T,C]
  flash_attn<<<dim3(B_DIM * NH_DIM * (T_DIM / 128)), 256, 0, stream>>>(qkvb, vtb, yb);

  // 5) output projection: [4096,1024] @ [1024,1024] -> f32 out
  gemm_bf16_wmma<false><<<dim3(C_DIM / 128, B_DIM * T_DIM / 128), 256, 0, stream>>>(
      yb, wprojb, out, B_DIM * T_DIM, C_DIM, C_DIM);
}
